// Trittention_pytorch_46823733461714
// MI455X (gfx1250) — compile-verified, matrix operand order checked by reading, not run on hardware
//
#include <hip/hip_runtime.h>
#include <hip/hip_bf16.h>

typedef __bf16 bf16_t;
typedef __attribute__((ext_vector_type(16))) __bf16 v16bf;
typedef __attribute__((ext_vector_type(8)))  __bf16 v8bf;
typedef __attribute__((ext_vector_type(8)))  float  v8f;

#define T_CTX 128
#define D_DIM 64
#define SCALE 0.125f
#define NEGF  -1000000.0f

// Order-preserving float -> u32 encoding (monotonic), so max over encodings
// == max over floats. Deterministic regardless of atomic ordering.
__device__ __forceinline__ unsigned fenc(float f) {
  int i = __float_as_int(f);
  return (i >= 0) ? ((unsigned)i | 0x80000000u) : ~(unsigned)i;
}
__device__ __forceinline__ float fdec(unsigned u) {
  int i = (u & 0x80000000u) ? (int)(u & 0x7fffffffu) : (int)~u;
  return __int_as_float(i);
}

__device__ __forceinline__ float silu(float x) { return x / (1.f + __expf(-x)); }

// WMMA bf16 operand fragment loader (16xK slice, K-step of 32).
// Matrix stored row-major in LDS ([M][K] for A-operand, [N][K] for B-operand).
// ISA layout: lane L (r=L&15, half=L>>4): elems 0..7 = K (k0 + half*8 .. +7),
// elems 8..15 = K (k0 + half*8 + 16 ..). Two ds_load_b128 per fragment.
__device__ __forceinline__ v16bf load_frag(const bf16_t* m, int row0, int stride, int k0) {
  int lane = threadIdx.x & 31;
  const bf16_t* p = m + (size_t)(row0 + (lane & 15)) * stride + k0 + ((lane >> 4) << 3);
  v8bf lo = *(const v8bf*)(p);
  v8bf hi = *(const v8bf*)(p + 16);
  return __builtin_shufflevector(lo, hi, 0, 1, 2, 3, 4, 5, 6, 7, 8, 9, 10, 11, 12, 13, 14, 15);
}

// ---------------------------------------------------------------------------
// Kernel 1: v_gated[b,h,p,t,:] = silu(v1[p,:]) * v2[t,:]   (bandwidth-bound)
// ---------------------------------------------------------------------------
__launch_bounds__(256)
__global__ void k_gate(const float* __restrict__ v1, const float* __restrict__ v2,
                       float* __restrict__ vg) {
  size_t g = (size_t)blockIdx.x * 256 + threadIdx.x;   // group of 4 elems
  int hg = (int)(g & 15);
  int tt = (int)((g >> 4) & 127);
  int p  = (int)((g >> 11) & 127);
  int bh = (int)(g >> 18);
  const float4 a4 = *(const float4*)(v1 + (size_t)(bh * 128 + p) * 64 + hg * 4);
  const float4 b4 = *(const float4*)(v2 + (size_t)(bh * 128 + tt) * 64 + hg * 4);
  float4 o;
  o.x = silu(a4.x) * b4.x;
  o.y = silu(a4.y) * b4.y;
  o.z = silu(a4.z) * b4.z;
  o.w = silu(a4.w) * b4.w;
  *(float4*)(vg + (size_t)((bh * 128 + p) * 128 + tt) * 64 + hg * 4) = o;
}

// ---------------------------------------------------------------------------
// Kernel 2: one WG per (b,h,j). S_j = (k1[j] (*) k2) @ q^T via bf16 WMMA,
// mask, write pre_softmax tile, atomic-max per-q scaled max into Menc.
// ---------------------------------------------------------------------------
__launch_bounds__(256)
__global__ void k_scores(const float* __restrict__ q, const float* __restrict__ k1,
                         const float* __restrict__ k2, float* __restrict__ pre,
                         unsigned* __restrict__ Menc) {
  __shared__ __align__(16) bf16_t lds_a[T_CTX * D_DIM];  // a[k][h] bf16
  __shared__ __align__(16) bf16_t lds_q[T_CTX * D_DIM];  // q[qq][h] bf16
  __shared__ float ldsK1[D_DIM];
  __shared__ unsigned ldsM[T_CTX];

  int t  = threadIdx.x;
  int j  = blockIdx.x & 127;
  int bh = blockIdx.x >> 7;
  const float* qb  = q  + (size_t)bh * T_CTX * D_DIM;
  const float* k1b = k1 + (size_t)bh * T_CTX * D_DIM;
  const float* k2b = k2 + (size_t)bh * T_CTX * D_DIM;

  if (t < D_DIM) ldsK1[t] = k1b[j * D_DIM + t];
  if (t < T_CTX) ldsM[t] = 0u;
  __syncthreads();

  for (int i = 0; i < 32; ++i) {                  // 8192 elements, coalesced
    int idx = t + 256 * i;
    int h = idx & 63;
    lds_a[idx] = (bf16_t)(ldsK1[h] * k2b[idx]);
    lds_q[idx] = (bf16_t)qb[idx];
  }
  __syncthreads();

  int w    = t >> 5;          // wave id == k-tile
  int lane = t & 31;
  int half = lane >> 4;

  v16bf a0 = load_frag(lds_a, w * 16, D_DIM, 0);
  v16bf a1 = load_frag(lds_a, w * 16, D_DIM, 32);
  float* preb = pre + (size_t)(bh * T_CTX + j) * (T_CTX * T_CTX);

  #pragma unroll
  for (int qt = 0; qt < 8; ++qt) {
    v16bf b0 = load_frag(lds_q, qt * 16, D_DIM, 0);
    v16bf b1 = load_frag(lds_q, qt * 16, D_DIM, 32);
    v8f acc = {};
    acc = __builtin_amdgcn_wmma_f32_16x16x32_bf16(false, a0, false, b0, (short)0, acc, false, false);
    acc = __builtin_amdgcn_wmma_f32_16x16x32_bf16(false, a1, false, b1, (short)0, acc, false, false);

    int qq = qt * 16 + (lane & 15);
    float vmax = -3.0e38f;
    #pragma unroll
    for (int i = 0; i < 8; ++i) {
      int kk = w * 16 + i + half * 8;
      bool m = (j > qq) | (kk > qq) | (qq > j + 128) | (qq > kk + 128) |
               (j + 128 < kk) | (kk + 128 < j);
      float val = m ? NEGF : acc[i];
      preb[(size_t)kk * T_CTX + qq] = val;
      vmax = fmaxf(vmax, val);
    }
    vmax = fmaxf(vmax, __shfl_xor(vmax, 16, 32));
    if (lane < 16) atomicMax(&ldsM[qq], fenc(vmax * SCALE));
  }
  __syncthreads();
  if (t < T_CTX) atomicMax(&Menc[bh * T_CTX + t], ldsM[t]);
}

// ---------------------------------------------------------------------------
// Kernel 3: one WG per (b,h,q-tile). Exact softmax (M known), accumulate
// z[q,h] = (1/L_true) * sum_j silu(v1[j,h]) * (P_j^T @ v2)[q,h] via bf16 WMMA.
// 128 threads = 4 waves, wave w handles h-tile w. The S_j tile (128x16 f32,
// 8KB) is double-buffered in LDS via GLOBAL_LOAD_ASYNC_TO_LDS_B128 so the
// next tile's HBM/L2 fetch overlaps the current tile's exp + WMMA work
// (tracked with ASYNCcnt / s_wait_asynccnt per CDNA5 ISA ch.10.7).
// ---------------------------------------------------------------------------
__launch_bounds__(128)
__global__ void k_softmax_z(const float* __restrict__ v1, const float* __restrict__ v2,
                            const float* __restrict__ pre, const unsigned* __restrict__ Menc,
                            float* __restrict__ zout, float* __restrict__ Mout,
                            float* __restrict__ Lout) {
  __shared__ __align__(16) float  Sbuf[2][T_CTX * 16];   // S_j column tile, double buffer
  __shared__ __align__(16) bf16_t ldsPT[16 * 136];       // P^T tile [q_local][k], pad 8
  __shared__ __align__(16) bf16_t v2T[D_DIM * T_CTX];    // [h][k]
  __shared__ float ldsSv[D_DIM];
  __shared__ float ldsLs[128];
  __shared__ float ldsLf[16];

  int t  = threadIdx.x;
  int qt = blockIdx.x & 7;
  int bh = blockIdx.x >> 3;
  const float* v1b  = v1 + (size_t)bh * T_CTX * D_DIM;
  const float* v2b  = v2 + (size_t)bh * T_CTX * D_DIM;
  const float* preb = pre + (size_t)bh * T_CTX * T_CTX * T_CTX;

  for (int i = 0; i < 64; ++i) {                  // build v2^T once (bf16)
    int idx = t + 128 * i;
    int h = idx & 63, k = idx >> 6;
    v2T[h * T_CTX + k] = (bf16_t)v2b[k * D_DIM + h];
  }

  int ql = t & 15;             // local q (phase A)
  int q  = qt * 16 + ql;
  int kb = t >> 4;             // 0..7
  float Msc  = fdec(Menc[bh * T_CTX + q]);
  float Lrun = 0.f;
  int lane = t & 31, half = lane >> 4, w = t >> 5;   // wave == h-tile
  v8f zacc = {};

  // Async-copy the 128x16 f32 column tile of S_j into LDS buffer `buf`.
  // 512 16-byte chunks / 128 threads = 4 issues per thread.
  auto issue_tile = [&](int j, int buf) {
    const float* src = preb + (size_t)j * (T_CTX * T_CTX) + qt * 16;
    #pragma unroll
    for (int i = 0; i < 4; ++i) {
      int c   = t + 128 * i;                 // chunk id
      int row = c >> 2, c4 = c & 3;          // row k, 16B sub-chunk
      unsigned loff = (unsigned)(uintptr_t)(&Sbuf[buf][row * 16 + c4 * 4]);
      const float* g = src + (size_t)row * T_CTX + c4 * 4;
      asm volatile("global_load_async_to_lds_b128 %0, %1, off"
                   :: "v"(loff), "v"(g) : "memory");
    }
  };

  issue_tile(0, 0);
  asm volatile("s_wait_asynccnt 0" ::: "memory");
  __syncthreads();                              // v2T build + first tile visible

  for (int j = 0; j < T_CTX; ++j) {
    int cur = j & 1;
    if (j + 1 < T_CTX) issue_tile(j + 1, cur ^ 1);   // overlap next fetch

    if (t < D_DIM) ldsSv[t] = silu(v1b[j * D_DIM + t]);
    #pragma unroll
    for (int i = 0; i < 16; ++i) {
      int k = kb + 8 * i;
      float s = Sbuf[cur][k * 16 + ql];
      float e = __expf(fmaf(s, SCALE, -Msc));   // masked -> exp(-125000-M)=0
      Lrun += e;
      ldsPT[ql * 136 + k] = (bf16_t)e;          // transposed bf16 store
    }
    __syncthreads();

    v8f g = {};
    #pragma unroll
    for (int ks = 0; ks < 4; ++ks) {
      v16bf a = load_frag(ldsPT, 0, 136, ks * 32);          // A = P^T rows (q)
      v16bf b = load_frag(v2T, w * 16, T_CTX, ks * 32);     // B = v2 cols (h)
      g = __builtin_amdgcn_wmma_f32_16x16x32_bf16(false, a, false, b, (short)0, g, false, false);
    }
    float sv = ldsSv[w * 16 + (lane & 15)];
    #pragma unroll
    for (int i = 0; i < 8; ++i) zacc[i] += g[i] * sv;

    asm volatile("s_wait_asynccnt 0" ::: "memory");  // own async chunks done
    __syncthreads();    // all waves' chunks visible + WAR guard on ldsPT/Sbuf
  }

  ldsLs[t] = Lrun;
  __syncthreads();
  if (t < 16) {
    float Lt = 0.f;
    #pragma unroll
    for (int g2 = 0; g2 < 8; ++g2) Lt += ldsLs[t + 16 * g2];
    ldsLf[t] = Lt;                                  // true softmax denominator
    Mout[bh * T_CTX + q] = Msc;
    Lout[bh * T_CTX + q] = Lt + 0.01f;
  }
  __syncthreads();
  #pragma unroll
  for (int i = 0; i < 8; ++i) {
    int m  = i + half * 8;
    int qg = qt * 16 + m;
    float z = zacc[i] / ldsLf[m];
    zout[(size_t)(bh * T_CTX + qg) * D_DIM + w * 16 + (lane & 15)] = z;
  }
}

// ---------------------------------------------------------------------------
extern "C" void kernel_launch(void* const* d_in, const int* in_sizes, int n_in,
                              void* d_out, int out_size, void* d_ws, size_t ws_size,
                              hipStream_t stream) {
  const float* q  = (const float*)d_in[0];
  const float* k1 = (const float*)d_in[1];
  const float* k2 = (const float*)d_in[2];
  const float* v1 = (const float*)d_in[3];
  const float* v2 = (const float*)d_in[4];
  float* out = (float*)d_out;

  const int BH = 32;  // B*H
  size_t off_pre = (size_t)BH * 128 * 64;                       // after z
  size_t off_vg  = off_pre + (size_t)BH * 128 * 128 * 128;      // after pre
  size_t off_M   = off_vg + (size_t)BH * 128 * 128 * 64;        // after v_gated
  size_t off_L   = off_M + (size_t)BH * 128;

  float* z_out = out;
  float* pre   = out + off_pre;
  float* vg    = out + off_vg;
  float* M_out = out + off_M;
  float* L_out = out + off_L;
  unsigned* Menc = (unsigned*)d_ws;                 // 16 KB of scratch

  hipMemsetAsync(Menc, 0, (size_t)BH * 128 * sizeof(unsigned), stream);
  k_gate<<<32768, 256, 0, stream>>>(v1, v2, vg);
  k_scores<<<4096, 256, 0, stream>>>(q, k1, k2, pre, Menc);
  k_softmax_z<<<256, 128, 0, stream>>>(v1, v2, pre, Menc, z_out, M_out, L_out);
}
